// EfficientRelationshipModel_77816217468973
// MI455X (gfx1250) — matrix operand order repaired and checked by reading, not compiled, not run on hardware
//
#include <hip/hip_runtime.h>
#include <hip/hip_bf16.h>

// EfficientRelationshipModel for MI455X (gfx1250, wave32).
// - Span MLPs evaluated only at the 512 gathered rows (8x FLOP cut vs ref).
// - All GEMMs via v_wmma_f32_16x16x32_bf16 (f32 accumulate).
// - Weights pre-packed into fragment-blocked bf16:
//     Bblk[ntile][kblock][lane][16]  -> wave B-fragment = 2 coalesced b128
//   loads at base+lane*32, streaming sequential 1 KB blocks per K-step.
// - A-tiles staged in LDS row-major; A-fragment = two ds_load_b128.
// - Dominant pairwise GEMM: one WG owns all 64 j-rows of a (b,i) pair
//   (64x2048 bf16 hidden tile, 256 KB LDS), 4Mx2N register blocking with
//   software-pipelined (double-buffered) B fragments: 512 WGs x 4 MB = 2 GB
//   total L2 B-traffic, 8 B global + 16 B LDS per lane per WMMA.

#define BB  8
#define LL  512
#define NN  64
#define DD  1024
#define DD2 2048

typedef __attribute__((ext_vector_type(16))) __bf16 v16bf;
typedef __attribute__((ext_vector_type(8)))  float  v8f;

union FragBF { v16bf v; uint4 q[2]; };

__device__ __forceinline__ v8f vzero() {
  v8f z = {0.f, 0.f, 0.f, 0.f, 0.f, 0.f, 0.f, 0.f};
  return z;
}

// 1M x 4N, software-pipelined B: one 16xK LDS A-tile vs 4 adjacent B tiles.
__device__ __forceinline__ void wave_gemm_1x4(const __hip_bfloat16* Alds,
                                              const __hip_bfloat16* __restrict__ Bblk,
                                              int K, int nt0, v8f acc[4]) {
  const int lane = threadIdx.x & 31;
  const int g    = lane >> 4;
  const int m    = lane & 15;
  const int kblocks = K >> 5;
  const __hip_bfloat16* arow  = Alds + (size_t)m * K;
  const __hip_bfloat16* bbase = Bblk + (size_t)nt0 * kblocks * 512 + lane * 16;
  FragBF bc[4];
#pragma unroll
  for (int t = 0; t < 4; ++t) {
    const uint4* bp = (const uint4*)(bbase + (size_t)t * kblocks * 512);
    bc[t].q[0] = bp[0];
    bc[t].q[1] = bp[1];
  }
  for (int kb = 0; kb < kblocks; ++kb) {
    FragBF a;
    a.q[0] = *(const uint4*)(arow + kb * 32 + g * 8);
    a.q[1] = *(const uint4*)(arow + kb * 32 + 16 + g * 8);
    const int kn = (kb + 1 < kblocks) ? (kb + 1) : kb;
    FragBF bn[4];
#pragma unroll
    for (int t = 0; t < 4; ++t) {
      const uint4* bp = (const uint4*)(bbase + ((size_t)t * kblocks + kn) * 512);
      bn[t].q[0] = bp[0];
      bn[t].q[1] = bp[1];
    }
    if ((kb & 3) == 0)
      __builtin_prefetch(bbase + (size_t)(kb + 8) * 512, 0, 1);
#pragma unroll
    for (int t = 0; t < 4; ++t)
      acc[t] = __builtin_amdgcn_wmma_f32_16x16x32_bf16(false, a.v, false, bc[t].v,
                                                       (short)0, acc[t], false, false);
#pragma unroll
    for (int t = 0; t < 4; ++t)
      bc[t] = bn[t];
  }
}

// 4M x 2N, software-pipelined B: four 16xK LDS A-tiles (rows m+16*mt) vs 2
// B tiles. acc index = mt*2 + nt.
__device__ __forceinline__ void wave_gemm_4x2(const __hip_bfloat16* Alds,
                                              const __hip_bfloat16* __restrict__ Bblk,
                                              int K, int nt0, v8f acc[8]) {
  const int lane = threadIdx.x & 31;
  const int g    = lane >> 4;
  const int m    = lane & 15;
  const int kblocks = K >> 5;
  const __hip_bfloat16* bbase = Bblk + (size_t)nt0 * kblocks * 512 + lane * 16;
  FragBF bc0, bc1;
  {
    const uint4* p0 = (const uint4*)bbase;
    const uint4* p1 = (const uint4*)(bbase + (size_t)kblocks * 512);
    bc0.q[0] = p0[0]; bc0.q[1] = p0[1];
    bc1.q[0] = p1[0]; bc1.q[1] = p1[1];
  }
  for (int kb = 0; kb < kblocks; ++kb) {
    const int kn = (kb + 1 < kblocks) ? (kb + 1) : kb;
    FragBF bn0, bn1;
    const uint4* p0 = (const uint4*)(bbase + (size_t)kn * 512);
    const uint4* p1 = (const uint4*)(bbase + ((size_t)kblocks + kn) * 512);
    bn0.q[0] = p0[0]; bn0.q[1] = p0[1];
    bn1.q[0] = p1[0]; bn1.q[1] = p1[1];
    if ((kb & 3) == 0) {
      __builtin_prefetch(bbase + (size_t)(kb + 16) * 512, 0, 1);
      __builtin_prefetch(bbase + ((size_t)kblocks + kb + 16) * 512, 0, 1);
    }
#pragma unroll
    for (int mt = 0; mt < 4; ++mt) {
      FragBF a;
      const __hip_bfloat16* arow = Alds + (size_t)(m + mt * 16) * K;
      a.q[0] = *(const uint4*)(arow + kb * 32 + g * 8);
      a.q[1] = *(const uint4*)(arow + kb * 32 + 16 + g * 8);
      acc[mt * 2 + 0] = __builtin_amdgcn_wmma_f32_16x16x32_bf16(
          false, a.v, false, bc0.v, (short)0, acc[mt * 2 + 0], false, false);
      acc[mt * 2 + 1] = __builtin_amdgcn_wmma_f32_16x16x32_bf16(
          false, a.v, false, bc1.v, (short)0, acc[mt * 2 + 1], false, false);
    }
    bc0 = bn0;
    bc1 = bn1;
  }
}

// ---------------------------------------------------------------------------
// Weight pack into fragment-blocked layout.
//   packed[((nt*kblocks + kb)*32 + lane)*16 + v] = bf16(W[k*N + n])
//   where n = nt*16 + (lane&15), k = kb*32 + (v>>3)*16 + (lane>>4)*8 + (v&7).
// ---------------------------------------------------------------------------
__global__ __launch_bounds__(256) void pack_w_kernel(const float* __restrict__ W,
                                                     __hip_bfloat16* __restrict__ out,
                                                     int K, int N) {
  __shared__ float tile[32][33];
  const int n0 = blockIdx.x * 32;
  const int k0 = blockIdx.y * 32;
  const int tx = threadIdx.x & 31;
  const int ty = threadIdx.x >> 5;
  for (int yy = ty; yy < 32; yy += 8)
    tile[yy][tx] = W[(size_t)(k0 + yy) * N + (n0 + tx)];
  __syncthreads();
  const int kblocks = K >> 5;
  const int kb = k0 >> 5;
  const int p   = threadIdx.x * 4;
  const int blk = p >> 9;
  const int l   = (p >> 4) & 31;
  const int v0  = p & 15;
  const int col  = l & 15;
  const int krow = ((l >> 4) << 3) + ((v0 & 8) << 1) + (v0 & 7);
  const int nt = (n0 >> 4) + blk;
  __hip_bfloat16* dst = out + ((size_t)nt * kblocks + kb) * 512 + l * 16 + v0;
#pragma unroll
  for (int u = 0; u < 4; ++u)
    dst[u] = __float2bfloat16(tile[krow + u][blk * 16 + col]);
}

// ---------------------------------------------------------------------------
// Fused gather + 2-layer MLP at the 512 gathered span rows.
// ---------------------------------------------------------------------------
__global__ __launch_bounds__(256) void span_mlp_kernel(
    const float* __restrict__ h, const long long* __restrict__ span, int which,
    const __hip_bfloat16* __restrict__ W1p, const float* __restrict__ b1,
    const __hip_bfloat16* __restrict__ W2p, const float* __restrict__ b2,
    float* __restrict__ rep) {
  extern __shared__ char smem[];
  __hip_bfloat16* Xt = (__hip_bfloat16*)smem;                    // 16 x 1024
  __hip_bfloat16* Tt = (__hip_bfloat16*)(smem + 16 * 1024 * 2);  // 16 x 2048
  const int tid = threadIdx.x;

  for (int e = tid * 4; e < 16 * DD; e += 1024) {
    const int r = e >> 10, c = e & (DD - 1);
    const int rg = blockIdx.x * 16 + r;
    const int b = rg >> 6, n = rg & 63;
    const long long idx = span[(size_t)(b * NN + n) * 2 + which];
    const float4 x = *(const float4*)(h + ((size_t)b * LL + (size_t)idx) * DD + c);
    Xt[e + 0] = __float2bfloat16(x.x);
    Xt[e + 1] = __float2bfloat16(x.y);
    Xt[e + 2] = __float2bfloat16(x.z);
    Xt[e + 3] = __float2bfloat16(x.w);
  }
  __syncthreads();

  const int wave = tid >> 5;
  const int lane = tid & 31;
  const int g = lane >> 4;
  const int m = lane & 15;

  // Layer 1: Tt = relu(X @ W1 + b1), 16 x 2048
  for (int grp = wave; grp < 32; grp += 8) {
    const int nt0 = grp * 4;
    v8f acc[4] = {vzero(), vzero(), vzero(), vzero()};
    wave_gemm_1x4(Xt, W1p, DD, nt0, acc);
#pragma unroll
    for (int t = 0; t < 4; ++t) {
      const int col = (nt0 + t) * 16 + m;
      const float bv = b1[col];
#pragma unroll
      for (int r = 0; r < 8; ++r) {
        float v = acc[t][r] + bv;
        v = v > 0.f ? v : 0.f;
        Tt[(size_t)(r + g * 8) * DD2 + col] = __float2bfloat16(v);
      }
    }
  }
  __syncthreads();

  // Layer 2: rep = Tt @ W2 + b2, 16 x 1024
  for (int grp = wave; grp < 16; grp += 8) {
    const int nt0 = grp * 4;
    v8f acc[4] = {vzero(), vzero(), vzero(), vzero()};
    wave_gemm_1x4(Tt, W2p, DD2, nt0, acc);
#pragma unroll
    for (int t = 0; t < 4; ++t) {
      const int col = (nt0 + t) * 16 + m;
      const float bv = b2[col];
#pragma unroll
      for (int r = 0; r < 8; ++r)
        rep[(size_t)(blockIdx.x * 16 + r + g * 8) * DD + col] = acc[t][r] + bv;
    }
  }
}

// ---------------------------------------------------------------------------
// entity = relu([start_rep, end_rep]) @ Wo + bo     (512 x 2048)
// ---------------------------------------------------------------------------
__global__ __launch_bounds__(256) void entity_kernel(
    const float* __restrict__ srep, const float* __restrict__ erep,
    const __hip_bfloat16* __restrict__ Wop, const float* __restrict__ bo,
    float* __restrict__ entity) {
  extern __shared__ char smem[];
  __hip_bfloat16* At = (__hip_bfloat16*)smem;  // 16 x 2048
  const int tid = threadIdx.x;
  for (int e = tid * 4; e < 16 * DD2; e += 1024) {
    const int r = e >> 11, c = e & (DD2 - 1);
    const int rg = blockIdx.x * 16 + r;
    const float* src = (c < DD) ? (srep + (size_t)rg * DD + c)
                                : (erep + (size_t)rg * DD + (c - DD));
    const float4 x = *(const float4*)src;
    At[e + 0] = __float2bfloat16(x.x > 0.f ? x.x : 0.f);
    At[e + 1] = __float2bfloat16(x.y > 0.f ? x.y : 0.f);
    At[e + 2] = __float2bfloat16(x.z > 0.f ? x.z : 0.f);
    At[e + 3] = __float2bfloat16(x.w > 0.f ? x.w : 0.f);
  }
  __syncthreads();
  const int wave = tid >> 5, lane = tid & 31, g = lane >> 4, m = lane & 15;
  for (int grp = wave; grp < 32; grp += 8) {
    const int nt0 = grp * 4;
    v8f acc[4] = {vzero(), vzero(), vzero(), vzero()};
    wave_gemm_1x4(At, Wop, DD2, nt0, acc);
#pragma unroll
    for (int t = 0; t < 4; ++t) {
      const int col = (nt0 + t) * 16 + m;
      const float bv = bo[col];
#pragma unroll
      for (int r = 0; r < 8; ++r)
        entity[(size_t)(blockIdx.x * 16 + r + g * 8) * DD2 + col] = acc[t][r] + bv;
    }
  }
}

// ---------------------------------------------------------------------------
// hi = entity @ Wr1[:2048] ; hj = entity @ Wr1[2048:]   (blockIdx.y selects)
// ---------------------------------------------------------------------------
__global__ __launch_bounds__(256) void pair_proj_kernel(
    const float* __restrict__ entity,
    const __hip_bfloat16* __restrict__ Wr1ap,
    const __hip_bfloat16* __restrict__ Wr1bp,
    float* __restrict__ hi, float* __restrict__ hj) {
  const __hip_bfloat16* Wp = blockIdx.y ? Wr1bp : Wr1ap;
  float* outp = blockIdx.y ? hj : hi;
  extern __shared__ char smem[];
  __hip_bfloat16* At = (__hip_bfloat16*)smem;  // 16 x 2048
  const int tid = threadIdx.x;
  for (int e = tid * 4; e < 16 * DD2; e += 1024) {
    const int r = e >> 11, c = e & (DD2 - 1);
    const int rg = blockIdx.x * 16 + r;
    const float4 x = *(const float4*)(entity + (size_t)rg * DD2 + c);
    At[e + 0] = __float2bfloat16(x.x);
    At[e + 1] = __float2bfloat16(x.y);
    At[e + 2] = __float2bfloat16(x.z);
    At[e + 3] = __float2bfloat16(x.w);
  }
  __syncthreads();
  const int wave = tid >> 5, lane = tid & 31, g = lane >> 4, m = lane & 15;
  for (int grp = wave; grp < 32; grp += 8) {
    const int nt0 = grp * 4;
    v8f acc[4] = {vzero(), vzero(), vzero(), vzero()};
    wave_gemm_1x4(At, Wp, DD2, nt0, acc);
#pragma unroll
    for (int t = 0; t < 4; ++t) {
      const int col = (nt0 + t) * 16 + m;
#pragma unroll
      for (int r = 0; r < 8; ++r)
        outp[(size_t)(blockIdx.x * 16 + r + g * 8) * DD2 + col] = acc[t][r];
    }
  }
}

// ---------------------------------------------------------------------------
// out[b,i,j,:] = relu(hi[b,i] + hj[b,j] + br1) @ Wr2 + br2
// One WG owns one (b,i): all 64 j-rows staged as a 64x2048 bf16 LDS tile
// (256 KB).  4Mx2N register blocking with double-buffered B fragments:
// per K-step 8 ds_load_b128 + 4 global b128 feed 8 WMMAs.  512 WGs.
// ---------------------------------------------------------------------------
__global__ __launch_bounds__(256) void relation_out_kernel(
    const float* __restrict__ hi, const float* __restrict__ hj,
    const float* __restrict__ br1,
    const __hip_bfloat16* __restrict__ Wr2p, const float* __restrict__ br2,
    float* __restrict__ out) {
  const int t = blockIdx.x;        // 0..511
  const int b = t >> 6;
  const int i = t & 63;
  extern __shared__ char smem[];
  __hip_bfloat16* Ht = (__hip_bfloat16*)smem;  // 64 x 2048
  const int tid = threadIdx.x;
  const float* hirow = hi + (size_t)(b * NN + i) * DD2;

  for (int e = tid * 4; e < 64 * DD2; e += 1024) {
    const int r = e >> 11, c = e & (DD2 - 1);        // r = j row 0..63
    const float4 a  = *(const float4*)(hirow + c);
    const float4 bj = *(const float4*)(hj + (size_t)(b * NN + r) * DD2 + c);
    const float4 bs = *(const float4*)(br1 + c);
    float v0 = a.x + bj.x + bs.x;
    float v1 = a.y + bj.y + bs.y;
    float v2 = a.z + bj.z + bs.z;
    float v3 = a.w + bj.w + bs.w;
    Ht[e + 0] = __float2bfloat16(v0 > 0.f ? v0 : 0.f);
    Ht[e + 1] = __float2bfloat16(v1 > 0.f ? v1 : 0.f);
    Ht[e + 2] = __float2bfloat16(v2 > 0.f ? v2 : 0.f);
    Ht[e + 3] = __float2bfloat16(v3 > 0.f ? v3 : 0.f);
  }
  __syncthreads();

  const int wave = tid >> 5, lane = tid & 31, g = lane >> 4, m = lane & 15;
  const size_t orow0 = (size_t)(b * NN + i) * NN * DD;
  // NT = 64 N-tiles as 32 pairs; 8 waves -> 4 passes of tile-pairs.
  for (int pr = wave; pr < 32; pr += 8) {
    const int nt0 = pr * 2;
    v8f acc[8] = {vzero(), vzero(), vzero(), vzero(),
                  vzero(), vzero(), vzero(), vzero()};  // [mt*2+nt]
    wave_gemm_4x2(Ht, Wr2p, DD2, nt0, acc);
#pragma unroll
    for (int mt = 0; mt < 4; ++mt) {
#pragma unroll
      for (int nt = 0; nt < 2; ++nt) {
        const int col = (nt0 + nt) * 16 + m;
        const float bv = br2[col];
        const v8f a = acc[mt * 2 + nt];
#pragma unroll
        for (int r = 0; r < 8; ++r)
          out[orow0 + (size_t)(mt * 16 + r + g * 8) * DD + col] = a[r] + bv;
      }
    }
  }
}

// ---------------------------------------------------------------------------
extern "C" void kernel_launch(void* const* d_in, const int* in_sizes, int n_in,
                              void* d_out, int out_size, void* d_ws, size_t ws_size,
                              hipStream_t stream) {
  const float*     h    = (const float*)d_in[0];
  const long long* span = (const long long*)d_in[1];  // reference dtype int64
  const float* Ws1 = (const float*)d_in[2];
  const float* bs1 = (const float*)d_in[3];
  const float* Ws2 = (const float*)d_in[4];
  const float* bs2 = (const float*)d_in[5];
  const float* We1 = (const float*)d_in[6];
  const float* be1 = (const float*)d_in[7];
  const float* We2 = (const float*)d_in[8];
  const float* be2 = (const float*)d_in[9];
  const float* Wo  = (const float*)d_in[10];
  const float* bo  = (const float*)d_in[11];
  const float* Wr1 = (const float*)d_in[12];
  const float* br1 = (const float*)d_in[13];
  const float* Wr2 = (const float*)d_in[14];
  const float* br2 = (const float*)d_in[15];
  float* out = (float*)d_out;

  // Workspace layout (~60 MB): packed bf16 weights then f32 intermediates.
  __hip_bfloat16* Ws1p  = (__hip_bfloat16*)d_ws;
  __hip_bfloat16* Ws2p  = Ws1p  + (size_t)DD  * DD2;
  __hip_bfloat16* We1p  = Ws2p  + (size_t)DD2 * DD;
  __hip_bfloat16* We2p  = We1p  + (size_t)DD  * DD2;
  __hip_bfloat16* Wop   = We2p  + (size_t)DD2 * DD;
  __hip_bfloat16* Wr1ap = Wop   + (size_t)DD2 * DD2;
  __hip_bfloat16* Wr1bp = Wr1ap + (size_t)DD2 * DD2;
  __hip_bfloat16* Wr2p  = Wr1bp + (size_t)DD2 * DD2;
  float* srep   = (float*)(Wr2p + (size_t)DD2 * DD);
  float* erep   = srep   + (size_t)BB * NN * DD;
  float* entity = erep   + (size_t)BB * NN * DD;
  float* hi     = entity + (size_t)BB * NN * DD2;
  float* hj     = hi     + (size_t)BB * NN * DD2;

  dim3 blk(256);

  // Weight packing (transpose + f32->bf16 + fragment blocking). grid=(N/32,K/32).
  pack_w_kernel<<<dim3(DD2 / 32, DD  / 32), blk, 0, stream>>>(Ws1, Ws1p, DD,  DD2);
  pack_w_kernel<<<dim3(DD  / 32, DD2 / 32), blk, 0, stream>>>(Ws2, Ws2p, DD2, DD);
  pack_w_kernel<<<dim3(DD2 / 32, DD  / 32), blk, 0, stream>>>(We1, We1p, DD,  DD2);
  pack_w_kernel<<<dim3(DD  / 32, DD2 / 32), blk, 0, stream>>>(We2, We2p, DD2, DD);
  pack_w_kernel<<<dim3(DD2 / 32, DD2 / 32), blk, 0, stream>>>(Wo,  Wop,  DD2, DD2);
  pack_w_kernel<<<dim3(DD2 / 32, DD2 / 32), blk, 0, stream>>>(Wr1, Wr1ap, DD2, DD2);
  pack_w_kernel<<<dim3(DD2 / 32, DD2 / 32), blk, 0, stream>>>(Wr1 + (size_t)DD2 * DD2, Wr1bp, DD2, DD2);
  pack_w_kernel<<<dim3(DD  / 32, DD2 / 32), blk, 0, stream>>>(Wr2, Wr2p, DD2, DD);

  // Span MLPs only at gathered positions (512 rows each).
  const size_t smem1 = (size_t)(16 * DD + 16 * DD2) * sizeof(__hip_bfloat16);  // 96 KB
  span_mlp_kernel<<<32, blk, smem1, stream>>>(h, span, 0, Ws1p, bs1, Ws2p, bs2, srep);
  span_mlp_kernel<<<32, blk, smem1, stream>>>(h, span, 1, We1p, be1, We2p, be2, erep);

  const size_t smem2 = (size_t)(16 * DD2) * sizeof(__hip_bfloat16);  // 64 KB
  entity_kernel<<<32, blk, smem2, stream>>>(srep, erep, Wop, bo, entity);
  pair_proj_kernel<<<dim3(32, 2), blk, smem2, stream>>>(entity, Wr1ap, Wr1bp, hi, hj);

  const size_t smem3 = (size_t)(64 * DD2) * sizeof(__hip_bfloat16);  // 256 KB
  relation_out_kernel<<<512, blk, smem3, stream>>>(hi, hj, br1, Wr2p, br2, out);
}
